// _DeepFMMachineModel_15401752724115
// MI455X (gfx1250) — matrix operand order repaired and checked by reading, not compiled, not run on hardware
//
#include <hip/hip_runtime.h>

// ---------------------------------------------------------------------------
// DeepFM on MI455X (gfx1250, wave32).  GEMMs via v_wmma_f32_16x16x32_f16.
// Problem is tiny (~270 MFLOP, <6MB) => L2-resident, latency bound.
// b1/b2 biases cancel exactly inside batchnorm (mean removes them) => skipped.
// ---------------------------------------------------------------------------

typedef __attribute__((ext_vector_type(16))) _Float16 v16h;
typedef __attribute__((ext_vector_type(8)))  _Float16 v8h;
typedef __attribute__((ext_vector_type(8)))  float    v8f;

#define BATCH   2048
#define NFIELD  8
#define FACTOR  16
#define D0      128     // NFIELD * FACTOR
#define H1DIM   256
#define H2DIM   128
#define EPSV    1e-5f

// ---------------------------------------------------------------------------
// Kernel 1: transpose + f32->f16 convert weights.
//   W1 [128][256] row-major  -> W1T [256][128]  (n-major, contiguous in K)
//   W2 [256][128] row-major  -> W2T [128][256]
// ---------------------------------------------------------------------------
__global__ void k_prep(const float* __restrict__ W1, const float* __restrict__ W2,
                       _Float16* __restrict__ W1T, _Float16* __restrict__ W2T) {
    int i = blockIdx.x * 256 + threadIdx.x;           // 0 .. 32767
    if (i < D0 * H1DIM) {
        int k = i / H1DIM, n = i % H1DIM;
        W1T[n * D0 + k] = (_Float16)W1[i];
    }
    if (i < H1DIM * H2DIM) {
        int k = i / H2DIM, n = i % H2DIM;
        W2T[n * H1DIM + k] = (_Float16)W2[i];
    }
}

// ---------------------------------------------------------------------------
// Kernel 2: embedding gather + FM first/second-order terms.
// One block (128 threads) per batch row. Thread t -> field t/16, dim t%16.
// Writes H as f16 (GEMM1 operand) and y_fm per row.
// ---------------------------------------------------------------------------
__global__ void k_gather(const int* __restrict__ x, const float* __restrict__ emb,
                         const float* __restrict__ wlin, const float* __restrict__ blin,
                         _Float16* __restrict__ Hh, float* __restrict__ yfm) {
    __shared__ float lds[D0];
    __shared__ float red[FACTOR + NFIELD];
    const int off[NFIELD] = {0, 100000, 150000, 151000, 151500, 151600, 151650, 151670};

    int b = blockIdx.x;
    int t = threadIdx.x;                // 0..127
    int f = t >> 4, d = t & 15;

    int idx   = x[b * NFIELD + f] + off[f];
    float val = emb[idx * FACTOR + d];
    lds[t] = val;
    Hh[b * D0 + t] = (_Float16)val;
    __syncthreads();

    if (t < FACTOR) {                   // per embedding-dim: s^2 - sum(v^2)
        float s = 0.f, ss = 0.f;
        #pragma unroll
        for (int ff = 0; ff < NFIELD; ++ff) {
            float v = lds[ff * FACTOR + t];
            s += v; ss += v * v;
        }
        red[t] = s * s - ss;
    } else if (t < FACTOR + NFIELD) {   // linear term gathers
        int ff = t - FACTOR;
        red[t] = wlin[x[b * NFIELD + ff] + off[ff]];
    }
    __syncthreads();

    if (t == 0) {
        float p = 0.f;
        #pragma unroll
        for (int i = 0; i < FACTOR; ++i) p += red[i];
        float l = 0.f;
        #pragma unroll
        for (int i = 0; i < NFIELD; ++i) l += red[FACTOR + i];
        yfm[b] = 0.5f * p + l + blin[0];
    }
}

// ---------------------------------------------------------------------------
// Kernel 3/5: WMMA GEMM.  C[M][N] = A[M][K] (f16 row-major) * BT[N][K] (f16
// n-major == B column-major).  One wave per 16x16 output tile, K,N compile-
// time so the K-loop fully unrolls into back-to-back v_wmma ops.
//
// A-fragment per CDNA5 ISA (16-bit A 16x32): lane L holds row
// M=L%16; halves 0..7 = K[hi*8 .. hi*8+8), halves 8..15 = K[16+hi*8 ..),
// hi = L/16  -> two contiguous 16B loads.
// B-fragment: lane L holds col N=L%16, same K order as A:
// halves 0..7 = K[hi*8 ..), halves 8..15 = K[16+hi*8 ..) -> two 16B loads.
// C/D: VGPR r -> row r + 8*hi, col L%16.
// ---------------------------------------------------------------------------
template <int K, int N>
__global__ void k_gemm(const _Float16* __restrict__ A, const _Float16* __restrict__ BT,
                       float* __restrict__ C) {
    int wave = (blockIdx.x * blockDim.x + threadIdx.x) >> 5;
    int lane = threadIdx.x & 31;
    constexpr int TN = N >> 4;
    int tm = wave / TN, tn = wave % TN;
    int ml = lane & 15, hi = lane >> 4;

    const _Float16* arow = A  + (tm * 16 + ml) * K;
    const _Float16* brow = BT + (tn * 16 + ml) * K;

    v8f c = {};
    #pragma unroll
    for (int k = 0; k < K; k += 32) {
        v8h a0 = *(const v8h*)(arow + k + hi * 8);
        v8h a1 = *(const v8h*)(arow + k + 16 + hi * 8);
        v8h b0 = *(const v8h*)(brow + k + hi * 8);
        v8h b1 = *(const v8h*)(brow + k + 16 + hi * 8);
        v16h a = __builtin_shufflevector(a0, a1, 0,1,2,3,4,5,6,7,8,9,10,11,12,13,14,15);
        v16h b = __builtin_shufflevector(b0, b1, 0,1,2,3,4,5,6,7,8,9,10,11,12,13,14,15);
        c = __builtin_amdgcn_wmma_f32_16x16x32_f16(false, a, false, b,
                                                   (short)0, c, false, false);
    }

    float* crow = C + (tm * 16 + hi * 8) * N + tn * 16 + ml;
    #pragma unroll
    for (int r = 0; r < 8; ++r) crow[r * N] = c[r];
}

// ---------------------------------------------------------------------------
// Kernel 4/6: fused batchnorm (batch statistics) + affine + ReLU.
// One block (256 threads) per output column; column fully reduced in-block.
// HALF_OUT=1 -> f16 output (GEMM operand), else f32.
// ---------------------------------------------------------------------------
template <int HALF_OUT, int N>
__global__ void k_bn_relu(const float* __restrict__ Cin, const float* __restrict__ g,
                          const float* __restrict__ beta, void* __restrict__ out) {
    __shared__ float rs[256], rss[256];
    int col = blockIdx.x, t = threadIdx.x;

    float s = 0.f, ss = 0.f;
    #pragma unroll
    for (int i = t; i < BATCH; i += 256) {
        float v = Cin[i * N + col];
        s += v; ss += v * v;
    }
    rs[t] = s; rss[t] = ss;
    __syncthreads();
    for (int st = 128; st > 0; st >>= 1) {
        if (t < st) { rs[t] += rs[t + st]; rss[t] += rss[t + st]; }
        __syncthreads();
    }
    float mean = rs[0] * (1.f / BATCH);
    float var  = rss[0] * (1.f / BATCH) - mean * mean;
    float gg   = g[col] * rsqrtf(var + EPSV);
    float bb   = beta[col] - mean * gg;

    #pragma unroll
    for (int i = t; i < BATCH; i += 256) {
        float o = fmaxf(Cin[i * N + col] * gg + bb, 0.f);
        if (HALF_OUT) ((_Float16*)out)[i * N + col] = (_Float16)o;
        else          ((float*)out)[i * N + col]    = o;
    }
}

// ---------------------------------------------------------------------------
// Kernel 7: y[b] = y_fm[b] + dot(H2[b,:], W3) + b3.  One block per row.
// ---------------------------------------------------------------------------
__global__ void k_final(const float* __restrict__ H2, const float* __restrict__ W3,
                        const float* __restrict__ b3, const float* __restrict__ yfm,
                        float* __restrict__ out) {
    __shared__ float r[H2DIM];
    int b = blockIdx.x, t = threadIdx.x;   // 128 threads
    r[t] = H2[b * H2DIM + t] * W3[t];
    __syncthreads();
    for (int st = 64; st > 0; st >>= 1) {
        if (t < st) r[t] += r[t + st];
        __syncthreads();
    }
    if (t == 0) out[b] = yfm[b] + r[0] + b3[0];
}

// ---------------------------------------------------------------------------
// Launch.  Inputs (setup_inputs order):
//  0:x 1:emb_table 2:w_lin 3:b_lin 4:W1 5:b1 6:g1 7:be1 8:W2 9:b2 10:g2
//  11:be2 12:W3 13:b3
// ---------------------------------------------------------------------------
extern "C" void kernel_launch(void* const* d_in, const int* in_sizes, int n_in,
                              void* d_out, int out_size, void* d_ws, size_t ws_size,
                              hipStream_t stream) {
    (void)in_sizes; (void)n_in; (void)out_size; (void)ws_size;

    const int*   x    = (const int*)  d_in[0];
    const float* emb  = (const float*)d_in[1];
    const float* wlin = (const float*)d_in[2];
    const float* blin = (const float*)d_in[3];
    const float* W1   = (const float*)d_in[4];
    const float* g1   = (const float*)d_in[6];
    const float* be1  = (const float*)d_in[7];
    const float* W2   = (const float*)d_in[8];
    const float* g2   = (const float*)d_in[10];
    const float* be2  = (const float*)d_in[11];
    const float* W3   = (const float*)d_in[12];
    const float* b3   = (const float*)d_in[13];
    float*       out  = (float*)d_out;

    // Workspace carve-up (all offsets 16B-aligned for b128 fragment loads).
    char* ws = (char*)d_ws;
    _Float16* Hh   = (_Float16*)(ws);                         // 2048*128*2  = 524288
    _Float16* W1T  = (_Float16*)(ws + 524288);                // 256*128*2   = 65536
    _Float16* W2T  = (_Float16*)(ws + 589824);                // 128*256*2   = 65536
    float*    yfm  = (float*)   (ws + 655360);                // 2048*4      = 8192
    float*    C1   = (float*)   (ws + 663552);                // 2048*256*4  = 2097152
    _Float16* H1h  = (_Float16*)(ws + 2760704);               // 2048*256*2  = 1048576
    float*    C2   = (float*)   (ws + 3809280);               // 2048*128*4  = 1048576
    float*    H2f  = (float*)   (ws + 4857856);               // 2048*128*4  = 1048576
                                                              // total 5906432 B

    // 1. weight transpose + f16 convert
    k_prep<<<128, 256, 0, stream>>>(W1, W2, W1T, W2T);

    // 2. embedding gather + FM terms
    k_gather<<<BATCH, D0, 0, stream>>>(x, emb, wlin, blin, Hh, yfm);

    // 3. GEMM1: (2048x128) x (128x256) -> C1. 2048 tiles, 8 waves/block.
    k_gemm<D0, H1DIM><<<256, 256, 0, stream>>>(Hh, W1T, C1);

    // 4. BN + ReLU -> H1 (f16)
    k_bn_relu<1, H1DIM><<<H1DIM, 256, 0, stream>>>(C1, g1, be1, (void*)H1h);

    // 5. GEMM2: (2048x256) x (256x128) -> C2. 1024 tiles.
    k_gemm<H1DIM, H2DIM><<<128, 256, 0, stream>>>(H1h, W2T, C2);

    // 6. BN + ReLU -> H2 (f32)
    k_bn_relu<0, H2DIM><<<H2DIM, 256, 0, stream>>>(C2, g2, be2, (void*)H2f);

    // 7. final dot + combine
    k_final<<<BATCH, H2DIM, 0, stream>>>(H2f, W3, b3, yfm, out);
}
